// SSDBlock_44427141710401
// MI455X (gfx1250) — compile-verified, matrix operand order checked by reading, not run on hardware
//
#include <hip/hip_runtime.h>
#include <cstdint>
#include <cstddef>

// ---------------------------------------------------------------------------
// Types for CDNA5 WMMA (wave32): v_wmma_f32_16x16x32_bf16
// ---------------------------------------------------------------------------
typedef __bf16 bf16_t;
typedef __attribute__((ext_vector_type(16))) __bf16 v16bf;
typedef __attribute__((ext_vector_type(8)))  float  v8f;
typedef __attribute__((ext_vector_type(4)))  int    v4i;

// Optional gfx1250 async global->LDS staging
#if defined(__gfx1250__) &&                                              \
    __has_builtin(__builtin_amdgcn_global_load_async_to_lds_b128) &&     \
    __has_builtin(__builtin_amdgcn_s_wait_asynccnt)
#define USE_ASYNC_LDS 1
#else
#define USE_ASYNC_LDS 0
#endif

#if USE_ASYNC_LDS
#define GLOBAL_AS __attribute__((address_space(1)))
#define LDS_AS    __attribute__((address_space(3)))
// Builtin param type (from compiler diagnostic): int4* (global), so cast via
// uintptr_t to AS-qualified int4 pointers (implicitly convertible to generic
// params as well on AMDGPU).
__device__ __forceinline__ void async_g2l_b128(const void* g, void* l) {
  __builtin_amdgcn_global_load_async_to_lds_b128(
      (GLOBAL_AS v4i*)(uintptr_t)g, (LDS_AS v4i*)(uintptr_t)l, 0, 0);
}
#endif

// ---------------------------------------------------------------------------
// Problem constants
// ---------------------------------------------------------------------------
constexpr int BATCH  = 2;
constexpr int LSEQ   = 2048;
constexpr int DMODEL = 2048;
constexpr int HEADS  = 32;
constexpr int NSTATE = 128;   // D_STATE
constexpr int HDIM   = 128;   // head dim
constexpr int DINNER = 4096;
constexpr int PROJD  = 16416; // 2*DINNER + 2*H*N + H
constexpr int CSZ    = 256;   // chunk size
constexpr int NCHUNK = LSEQ / CSZ;
constexpr int GOFF   = DINNER;                      // gate columns
constexpr int BOFF   = 2 * DINNER;                  // B columns
constexpr int COFF   = 2 * DINNER + HEADS * NSTATE; // C columns
constexpr int DTOFF  = 2 * DINNER + 2 * HEADS * NSTATE; // dt columns

// SSD kernel LDS layout (bytes)
constexpr int HPITCH  = 132; // f32 elems per h row (padded)
constexpr int HBPITCH = 136; // bf16 elems per h-mirror row (padded, 272B rows)
constexpr int TPITCH  = 264; // bf16 elems per transposed row (padded)
constexpr int MPITCH  = 136; // bf16 elems per M row (padded)
constexpr int SSD_LDS_BYTES =
    128 * HPITCH * 4 +       // h state (f32)            67584
    128 * TPITCH * 2 +       // val^T (bf16)             67584
    128 * TPITCH * 2 +       // B^T   (bf16)             67584
    CSZ * MPITCH * 2 +       // M half (bf16)            69632
    CSZ * 4 +                // cum (f32)
    CSZ * 4 +                // w = exp(last-cum)
    128 * HBPITCH * 2;       // h mirror (bf16)          34816
static_assert(SSD_LDS_BYTES <= 320 * 1024, "LDS over WGP budget");

// ---------------------------------------------------------------------------
// Helpers
// ---------------------------------------------------------------------------
__device__ __forceinline__ v8f zero_v8f() {
  v8f z;
#pragma unroll
  for (int i = 0; i < 8; ++i) z[i] = 0.0f;
  return z;
}

// A/B fragment = 16 bf16 per lane: {k0..k0+7, k0+16..k0+23}
__device__ __forceinline__ v16bf load2x8(const bf16_t* p0, const bf16_t* p1) {
  union { v16bf v; uint4 q[2]; } u;
  u.q[0] = *reinterpret_cast<const uint4*>(p0);
  u.q[1] = *reinterpret_cast<const uint4*>(p1);
  return u.v;
}

__device__ __forceinline__ v8f wmma_bf16(v16bf a, v16bf b, v8f c) {
  return __builtin_amdgcn_wmma_f32_16x16x32_bf16(false, a, false, b, (short)0, c,
                                                 false, false);
}

__device__ __forceinline__ float silu_f(float x) {
  return x * (1.0f / (1.0f + __expf(-x)));
}

// ---------------------------------------------------------------------------
// fp32 -> bf16 conversion
// ---------------------------------------------------------------------------
__global__ __launch_bounds__(256) void cvt_kernel(const float* __restrict__ s,
                                                  bf16_t* __restrict__ d,
                                                  size_t n) {
  size_t i = (size_t)blockIdx.x * 256 + threadIdx.x;
  if (i < n) d[i] = (bf16_t)s[i];
}

// ---------------------------------------------------------------------------
// Generic bf16 GEMM: C[M,N](OutT) = A[M,K] @ W[N,K]^T, all row-major bf16.
// 128x128 block tile, K-step 32, 8 waves of (2x4) 16x16 WMMA tiles.
// ---------------------------------------------------------------------------
template <typename OutT>
__global__ __launch_bounds__(256) void gemm_bf16_kernel(
    const bf16_t* __restrict__ A, const bf16_t* __restrict__ W,
    OutT* __restrict__ C, int M, int N, int K) {
  constexpr int TP = 40;  // padded bf16 row pitch (20 dwords -> conflict free)
  __shared__ bf16_t sA[128 * TP];
  __shared__ bf16_t sB[128 * TP];

  const int tid  = threadIdx.x;
  const int lane = tid & 31;
  const int wv   = tid >> 5;       // 8 waves
  const int wm   = wv >> 1;        // 4 along M (32 rows)
  const int wn   = wv & 1;         // 2 along N (64 cols)
  const int lr   = lane & 15;
  const int lkb  = (lane >> 4) * 8;
  const int orow = (lane >> 4) * 8;
  const int bm   = blockIdx.y * 128;
  const int bn   = blockIdx.x * 128;

  v8f acc[2][4];
#pragma unroll
  for (int i = 0; i < 2; ++i)
#pragma unroll
    for (int j = 0; j < 4; ++j) acc[i][j] = zero_v8f();

  // Each thread stages two 8-element chunks per tile per matrix.
  const int r0 = tid >> 2,           c0 = (tid & 3) * 8;
  const int r1 = (tid + 256) >> 2,   c1 = ((tid + 256) & 3) * 8;

  for (int kb = 0; kb < K; kb += 32) {
    {
      int wr0 = bn + r0; if (wr0 >= N) wr0 = N - 1;  // N-tail clamp
      int wr1 = bn + r1; if (wr1 >= N) wr1 = N - 1;
#if USE_ASYNC_LDS
      // gfx1250 async copy: 16B per lane, tracked by ASYNCcnt
      async_g2l_b128(A + (size_t)(bm + r0) * K + kb + c0, sA + r0 * TP + c0);
      async_g2l_b128(A + (size_t)(bm + r1) * K + kb + c1, sA + r1 * TP + c1);
      async_g2l_b128(W + (size_t)wr0 * K + kb + c0,       sB + r0 * TP + c0);
      async_g2l_b128(W + (size_t)wr1 * K + kb + c1,       sB + r1 * TP + c1);
      __builtin_amdgcn_s_wait_asynccnt(0);
#else
      uint4 a0 = *(const uint4*)(A + (size_t)(bm + r0) * K + kb + c0);
      uint4 a1 = *(const uint4*)(A + (size_t)(bm + r1) * K + kb + c1);
      uint4 b0 = *(const uint4*)(W + (size_t)wr0 * K + kb + c0);
      uint4 b1 = *(const uint4*)(W + (size_t)wr1 * K + kb + c1);
      *(uint4*)(sA + r0 * TP + c0) = a0;
      *(uint4*)(sA + r1 * TP + c1) = a1;
      *(uint4*)(sB + r0 * TP + c0) = b0;
      *(uint4*)(sB + r1 * TP + c1) = b1;
#endif
      if (kb + 32 < K) {  // warm next slab (global_prefetch_b8)
        __builtin_prefetch(A + (size_t)(bm + r0) * K + kb + 32 + c0, 0, 1);
        __builtin_prefetch(W + (size_t)wr0 * K + kb + 32 + c0, 0, 1);
      }
    }
    __syncthreads();

    v16bf af[2];
#pragma unroll
    for (int mt = 0; mt < 2; ++mt) {
      const bf16_t* p = sA + (wm * 32 + mt * 16 + lr) * TP + lkb;
      af[mt] = load2x8(p, p + 16);
    }
#pragma unroll
    for (int nt = 0; nt < 4; ++nt) {
      const bf16_t* p = sB + (wn * 64 + nt * 16 + lr) * TP + lkb;
      v16bf bfr = load2x8(p, p + 16);
#pragma unroll
      for (int mt = 0; mt < 2; ++mt)
        acc[mt][nt] = wmma_bf16(af[mt], bfr, acc[mt][nt]);
    }
    __syncthreads();
  }

#pragma unroll
  for (int mt = 0; mt < 2; ++mt)
#pragma unroll
    for (int nt = 0; nt < 4; ++nt)
#pragma unroll
      for (int r = 0; r < 8; ++r) {
        int grow = bm + wm * 32 + mt * 16 + r + orow;
        int gcol = bn + wn * 64 + nt * 16 + lr;
        if (gcol < N) C[(size_t)grow * N + gcol] = (OutT)acc[mt][nt][r];
      }
}

// ---------------------------------------------------------------------------
// ld[b,l,h] = -exp(A_log[h]) * softplus(dt + dt_bias[h])
// ---------------------------------------------------------------------------
__global__ __launch_bounds__(256) void ld_kernel(
    const bf16_t* __restrict__ projb, const float* __restrict__ A_log,
    const float* __restrict__ dt_bias, float* __restrict__ ld) {
  size_t idx = (size_t)blockIdx.x * 256 + threadIdx.x;  // over B*L*H
  int h = (int)(idx % HEADS);
  size_t bl = idx / HEADS;
  float x = (float)projb[bl * PROJD + DTOFF + h] + dt_bias[h];
  float sp = (x > 20.0f) ? x : log1pf(__expf(x));
  ld[idx] = -__expf(A_log[h]) * sp;
}

// ---------------------------------------------------------------------------
// Depthwise causal conv (D_CONV=4) + bias + SiLU  -> bf16
// ---------------------------------------------------------------------------
__global__ __launch_bounds__(256) void conv_silu_kernel(
    const bf16_t* __restrict__ projb, const float* __restrict__ conv_w,
    const float* __restrict__ conv_b, bf16_t* __restrict__ valc) {
  size_t idx = (size_t)blockIdx.x * 256 + threadIdx.x;  // over B*L*DINNER
  int ch = (int)(idx % DINNER);
  size_t bl = idx / DINNER;
  int l = (int)(bl % LSEQ);
  int b = (int)(bl / LSEQ);
  float s = conv_b[ch];
#pragma unroll
  for (int j = 0; j < 4; ++j) {
    int li = l + j - 3;
    if (li >= 0)
      s += conv_w[ch * 4 + j] *
           (float)projb[((size_t)b * LSEQ + li) * PROJD + ch];
  }
  valc[idx] = (bf16_t)silu_f(s);
}

// ---------------------------------------------------------------------------
// SSD chunked scan. One block per (b,h); 512 threads = 16 waves; 8 chunks
// processed sequentially with state h[128][128] resident in LDS: f32 master
// for accumulation + bf16 mirror so the C@h^T stage feeds WMMA with straight
// ds_load_b128 fragments (no serialized f32->bf16 cvt chains).
// ---------------------------------------------------------------------------
__global__ __launch_bounds__(512) void ssd_kernel(
    const bf16_t* __restrict__ projb,   // [B*L][PROJD]   (B/C columns)
    const bf16_t* __restrict__ valc,    // [B*L][DINNER]  conv'd values
    const float*  __restrict__ ld,      // [B*L][H]       decay log
    const float*  __restrict__ Dp,      // [DINNER]
    bf16_t* __restrict__ Yb)            // [B*L][DINNER]  output (pre-gate)
{
  extern __shared__ char smem[];
  float*  sh   = (float*)smem;                               // [128][HPITCH]
  bf16_t* svT  = (bf16_t*)(smem + 128 * HPITCH * 4);         // [128][TPITCH]
  bf16_t* sBT  = svT + 128 * TPITCH;                         // [128][TPITCH]
  bf16_t* sM   = sBT + 128 * TPITCH;                         // [CSZ][MPITCH]
  float*  scum = (float*)(sM + CSZ * MPITCH);                // [CSZ]
  float*  sw   = scum + CSZ;                                 // [CSZ]
  bf16_t* shb  = (bf16_t*)(sw + CSZ);                        // [128][HBPITCH]

  const int bh   = blockIdx.x;        // 0..63
  const int bIdx = bh >> 5;           // batch
  const int h    = bh & 31;           // head
  const int tid  = threadIdx.x;
  const int lane = tid & 31;
  const int wv   = tid >> 5;          // 16 waves
  const int lr   = lane & 15;
  const int lkb  = (lane >> 4) * 8;
  const int orow = (lane >> 4) * 8;
  const size_t rowBase = (size_t)bIdx * LSEQ;

  for (int i = tid; i < 128 * HPITCH; i += 512) sh[i] = 0.0f;
  for (int i = tid; i < 128 * HBPITCH; i += 512) shb[i] = (bf16_t)0.0f;
  __syncthreads();

  for (int c = 0; c < NCHUNK; ++c) {
    const int l0 = c * CSZ;

    // ---- stage val^T, B^T, and raw decay ----
    {
      int row  = tid >> 1;   // 0..255 chunk row
      int half = tid & 1;    // which 64-wide slab of d/n
      const bf16_t* vp = valc + (rowBase + l0 + row) * (size_t)DINNER +
                         h * HDIM + half * 64;
      const bf16_t* bp = projb + (rowBase + l0 + row) * (size_t)PROJD +
                         BOFF + h * NSTATE + half * 64;
#pragma unroll 8
      for (int j = 0; j < 64; ++j) {
        svT[(half * 64 + j) * TPITCH + row] = vp[j];
        sBT[(half * 64 + j) * TPITCH + row] = bp[j];
      }
      if (tid < CSZ) scum[tid] = ld[(rowBase + l0 + tid) * HEADS + h];
    }
    __syncthreads();

    // ---- inclusive scan of decay over the chunk ----
    for (int off = 1; off < CSZ; off <<= 1) {
      float v = 0.0f;
      if (tid < CSZ && tid >= off) v = scum[tid - off];
      __syncthreads();
      if (tid < CSZ && tid >= off) scum[tid] += v;
      __syncthreads();
    }
    if (tid < CSZ) sw[tid] = __expf(scum[CSZ - 1] - scum[tid]);
    __syncthreads();

    const int lt = wv;  // each wave owns one 16-row l-tile
    const bf16_t* Cbase = projb + (rowBase + l0) * (size_t)PROJD + COFF + h * NSTATE;
    const bf16_t* Bbase = projb + (rowBase + l0) * (size_t)PROJD + BOFF + h * NSTATE;

    // C-row fragments (A operand for both CB and C@h^T), K over NSTATE
    v16bf afrC[4];
    {
      const bf16_t* arow = Cbase + (size_t)(lt * 16 + lr) * PROJD;
#pragma unroll
      for (int ks = 0; ks < 4; ++ks)
        afrC[ks] = load2x8(arow + ks * 32 + lkb, arow + ks * 32 + lkb + 16);
    }

    v8f Yacc[8];
#pragma unroll
    for (int dt = 0; dt < 8; ++dt) Yacc[dt] = zero_v8f();

    // ---- inter-chunk: Y += exp(cum[l]) * (C @ h^T), h from bf16 mirror ----
    for (int dt = 0; dt < 8; ++dt) {
      v8f acc = zero_v8f();
#pragma unroll
      for (int ks = 0; ks < 4; ++ks) {
        const bf16_t* hrow = shb + (dt * 16 + lr) * HBPITCH + ks * 32 + lkb;
        acc = wmma_bf16(afrC[ks], load2x8(hrow, hrow + 16), acc);
      }
#pragma unroll
      for (int r = 0; r < 8; ++r) {
        int lrow = lt * 16 + r + orow;
        Yacc[dt][r] += __expf(scum[lrow]) * acc[r];
      }
    }

    // ---- intra-chunk in two 128-column halves of M ----
    for (int half = 0; half < 2; ++half) {
      // M_half[l][m] = (m<=l) ? (C@B^T)[l][m] * exp(cum[l]-cum[m]) : 0
      for (int mt = 0; mt < 8; ++mt) {
        v8f acc = zero_v8f();
        const bf16_t* brow = Bbase + (size_t)(half * 128 + mt * 16 + lr) * PROJD;
#pragma unroll
        for (int ks = 0; ks < 4; ++ks)
          acc = wmma_bf16(afrC[ks],
                          load2x8(brow + ks * 32 + lkb, brow + ks * 32 + lkb + 16),
                          acc);
#pragma unroll
        for (int r = 0; r < 8; ++r) {
          int lrow = lt * 16 + r + orow;
          int mcol = half * 128 + mt * 16 + lr;
          float v = (mcol <= lrow)
                        ? acc[r] * __expf(scum[lrow] - scum[mcol])
                        : 0.0f;
          sM[lrow * MPITCH + mt * 16 + lr] = (bf16_t)v;
        }
      }
      __syncthreads();

      // Y += M_half @ val[half-rows]
      {
        v16bf afrM[4];
#pragma unroll
        for (int ks = 0; ks < 4; ++ks) {
          const bf16_t* arow = sM + (lt * 16 + lr) * MPITCH + ks * 32 + lkb;
          afrM[ks] = load2x8(arow, arow + 16);
        }
        for (int dt = 0; dt < 8; ++dt) {
          v8f acc = Yacc[dt];
#pragma unroll
          for (int ks = 0; ks < 4; ++ks) {
            const bf16_t* brow =
                svT + (dt * 16 + lr) * TPITCH + half * 128 + ks * 32 + lkb;
            acc = wmma_bf16(afrM[ks], load2x8(brow, brow + 16), acc);
          }
          Yacc[dt] = acc;
        }
      }
      __syncthreads();
    }

    // ---- D*val skip term + store Y (bf16) ----
    for (int dt = 0; dt < 8; ++dt) {
#pragma unroll
      for (int r = 0; r < 8; ++r) {
        int lrow = lt * 16 + r + orow;
        int d = dt * 16 + lr;
        float v = Yacc[dt][r] +
                  Dp[h * HDIM + d] * (float)svT[d * TPITCH + lrow];
        Yb[(rowBase + l0 + lrow) * (size_t)DINNER + h * HDIM + d] = (bf16_t)v;
      }
    }

    // ---- state update: h = exp(last)*h + (val*w)^T @ B  (f32 + bf16 mirror)
    {
      const int dt = wv >> 1;                  // 8 d-tiles, 2 waves each
      const float hscale = __expf(scum[CSZ - 1]);
      for (int j = 0; j < 4; ++j) {
        const int nt = (wv & 1) * 4 + j;
        v8f acc = zero_v8f();
#pragma unroll
        for (int ks = 0; ks < 8; ++ks) {       // K = 256 chunk rows
          const int d = dt * 16 + lr;
          const bf16_t* vr = svT + d * TPITCH + ks * 32 + lkb;
          const float*  wr = sw + ks * 32 + lkb;
          v16bf afr;
#pragma unroll
          for (int i = 0; i < 8; ++i) {
            afr[i]     = (bf16_t)((float)vr[i]      * wr[i]);
            afr[i + 8] = (bf16_t)((float)vr[16 + i] * wr[16 + i]);
          }
          const bf16_t* br = sBT + (nt * 16 + lr) * TPITCH + ks * 32 + lkb;
          acc = wmma_bf16(afr, load2x8(br, br + 16), acc);
        }
#pragma unroll
        for (int r = 0; r < 8; ++r) {
          int d = dt * 16 + r + orow;
          int n = nt * 16 + lr;
          float nv = sh[d * HPITCH + n] * hscale + acc[r];
          sh[d * HPITCH + n] = nv;
          shb[d * HBPITCH + n] = (bf16_t)nv;
        }
      }
    }
    __syncthreads();
  }
}

// ---------------------------------------------------------------------------
// Y *= silu(gate)   (in place on bf16 Y)
// ---------------------------------------------------------------------------
__global__ __launch_bounds__(256) void gate_kernel(
    const bf16_t* __restrict__ projb, bf16_t* __restrict__ Yb) {
  size_t idx = (size_t)blockIdx.x * 256 + threadIdx.x;  // over B*L*DINNER
  int ch = (int)(idx % DINNER);
  size_t bl = idx / DINNER;
  float g = (float)projb[bl * PROJD + GOFF + ch];
  Yb[idx] = (bf16_t)((float)Yb[idx] * silu_f(g));
}

// ---------------------------------------------------------------------------
// Host orchestration
// ---------------------------------------------------------------------------
extern "C" void kernel_launch(void* const* d_in, const int* in_sizes, int n_in,
                              void* d_out, int out_size, void* d_ws,
                              size_t ws_size, hipStream_t stream) {
  (void)in_sizes; (void)n_in; (void)out_size; (void)ws_size;
  const float* x      = (const float*)d_in[0];
  const float* W_in   = (const float*)d_in[1];
  const float* conv_w = (const float*)d_in[2];
  const float* conv_b = (const float*)d_in[3];
  const float* A_log  = (const float*)d_in[4];
  const float* D_par  = (const float*)d_in[5];
  const float* dt_b   = (const float*)d_in[6];
  const float* W_out  = (const float*)d_in[7];
  float* out = (float*)d_out;

  char* ws = (char*)d_ws;
  size_t off = 0;
  auto carve = [&](size_t bytes) -> void* {
    void* p = ws + off;
    off += (bytes + 255) & ~(size_t)255;
    return p;
  };
  bf16_t* xbf   = (bf16_t*)carve((size_t)BATCH * LSEQ * DMODEL * 2);
  bf16_t* Winb  = (bf16_t*)carve((size_t)PROJD * DMODEL * 2);
  bf16_t* Woutb = (bf16_t*)carve((size_t)DMODEL * DINNER * 2);
  bf16_t* projb = (bf16_t*)carve((size_t)BATCH * LSEQ * PROJD * 2);
  bf16_t* valcb = (bf16_t*)carve((size_t)BATCH * LSEQ * DINNER * 2);
  float*  ldbuf = (float*) carve((size_t)BATCH * LSEQ * HEADS * 4);
  bf16_t* Ybf   = (bf16_t*)carve((size_t)BATCH * LSEQ * DINNER * 2);

  // fp32 -> bf16 once
  cvt_kernel<<<(BATCH * LSEQ * DMODEL) / 256, 256, 0, stream>>>(
      x, xbf, (size_t)BATCH * LSEQ * DMODEL);
  cvt_kernel<<<(PROJD * DMODEL) / 256, 256, 0, stream>>>(
      W_in, Winb, (size_t)PROJD * DMODEL);
  cvt_kernel<<<(DMODEL * DINNER) / 256, 256, 0, stream>>>(
      W_out, Woutb, (size_t)DMODEL * DINNER);

  // input projection: [4096,16416] = x[4096,2048] @ W_in^T
  {
    dim3 g((PROJD + 127) / 128, (BATCH * LSEQ) / 128);
    gemm_bf16_kernel<bf16_t><<<g, 256, 0, stream>>>(
        xbf, Winb, projb, BATCH * LSEQ, PROJD, DMODEL);
  }

  ld_kernel<<<(BATCH * LSEQ * HEADS) / 256, 256, 0, stream>>>(
      projb, A_log, dt_b, ldbuf);
  conv_silu_kernel<<<(BATCH * LSEQ * DINNER) / 256, 256, 0, stream>>>(
      projb, conv_w, conv_b, valcb);

  // SSD scan: 64 blocks (b,h) x 512 threads, ~302KB dynamic LDS
  (void)hipFuncSetAttribute((const void*)ssd_kernel,
                            hipFuncAttributeMaxDynamicSharedMemorySize,
                            SSD_LDS_BYTES);
  ssd_kernel<<<BATCH * HEADS, 512, SSD_LDS_BYTES, stream>>>(
      projb, valcb, ldbuf, D_par, Ybf);

  gate_kernel<<<(BATCH * LSEQ * DINNER) / 256, 256, 0, stream>>>(projb, Ybf);

  // output projection: [4096,2048] = Z[4096,4096] @ W_out^T  -> fp32 d_out
  {
    dim3 g(DMODEL / 128, (BATCH * LSEQ) / 128);
    gemm_bf16_kernel<float><<<g, 256, 0, stream>>>(
        Ybf, Woutb, out, BATCH * LSEQ, DMODEL, DINNER);
  }
}
